// PagedSdpaAttention_89618787598500
// MI455X (gfx1250) — matrix operand order repaired and checked by reading, not compiled, not run on hardware
//
#include <hip/hip_runtime.h>
#include <hip/hip_bf16.h>
#include <stdint.h>

// ---------------------------------------------------------------------------
// PagedSdpaAttention for MI455X (gfx1250): bf16 WMMA + async-to-LDS pipeline.
//   K1: q = hidden(512x4096) @ Wq(4096x4096)            [WMMA GEMM, async DB]
//   K2: RoPE + repack q -> bf16 [32 heads][512][128]
//   K3: flash attention, GQA-aware (1 WG = 1 kv-head x 4 q-heads) [WMMA, async DB]
//   K4: out = ctx(512x4096) @ Wo(4096x4096)             [WMMA GEMM, async DB]
// ---------------------------------------------------------------------------

typedef __attribute__((ext_vector_type(16))) __bf16 v16bf;
typedef __attribute__((ext_vector_type(8)))  float  v8f;

union BF16x16 { v16bf v; uint4 q[2]; unsigned u[8]; unsigned short s[16]; };
union F32x8   { v8f   v; float f[8]; };

__device__ __forceinline__ unsigned short f2bf(float x) {
    unsigned u = __builtin_bit_cast(unsigned, x);
    u += 0x7FFFu + ((u >> 16) & 1u);          // round-to-nearest-even
    return (unsigned short)(u >> 16);
}
__device__ __forceinline__ unsigned pack2(float a, float b) {
    return (unsigned)f2bf(a) | ((unsigned)f2bf(b) << 16);
}

// gfx1250 async copy: global -> LDS, 16B per lane, tracked by ASYNCcnt.
__device__ __forceinline__ void async_ld16(unsigned ldsoff, const float* g) {
    asm volatile("global_load_async_to_lds_b128 %0, %1, off"
                 :: "v"(ldsoff), "v"(g) : "memory");
}
__device__ __forceinline__ unsigned lds_off(const void* p) {
    return (unsigned)(uintptr_t)p;   // LDS aperture: addr[31:0] == LDS offset
}
#define WAIT_ASYNC(n) asm volatile("s_wait_asynccnt %0" :: "n"(n) : "memory")

#define S_TOT   512
#define HID     4096
#define NHEADS  32
#define NKVH    8
#define HD      128
#define NCHUNK  256   // 8192 keys / 32 per chunk

// ---------------------------------------------------------------------------
// K1/K4: C(MxN) = A(MxK,f32,row) @ B(KxN,f32,row), bf16 WMMA, f32 out.
// WG = 256 thr (8 waves), tile 64x128, K-step 32, async double-buffered.
// ---------------------------------------------------------------------------
__global__ __launch_bounds__(256)
void psdpa_gemm_bf16(const float* __restrict__ A, const float* __restrict__ B,
                     float* __restrict__ C, int M, int N, int K) {
    __shared__ __align__(16) float          af32[2][64][32];   // raw A tiles
    __shared__ __align__(16) float          bf32[2][32][128];  // raw B tiles
    __shared__ __align__(16) unsigned       lds_a[64][16];     // 64x32 bf16 row-major
    __shared__ __align__(16) unsigned short lds_bt[128][32];   // 32x128 bf16 transposed

    const int tid  = threadIdx.x;
    const int lane = tid & 31;
    const int wave = tid >> 5;
    const bool lo  = lane < 16;
    const int gm0  = blockIdx.y * 64;
    const int gn0  = blockIdx.x * 128;
    const int m0   = (wave & 3) * 16;
    const int n0   = (wave >> 2) * 64;

    F32x8 acc[4];
#pragma unroll
    for (int i = 0; i < 4; ++i)
#pragma unroll
        for (int r = 0; r < 8; ++r) acc[i].f[r] = 0.0f;

    // issue tile for K-offset kt into fp32 buffer b (6 async instrs / thread)
    auto issue = [&](int kt, int b) {
#pragma unroll
        for (int it = 0; it < 2; ++it) {
            int idx = tid + 256 * it;            // 0..511 x 16B (A: 64x32)
            int row = idx >> 3, c4 = idx & 7;
            async_ld16(lds_off(&af32[b][row][c4 * 4]),
                       A + (size_t)(gm0 + row) * K + kt + c4 * 4);
        }
#pragma unroll
        for (int it = 0; it < 4; ++it) {
            int idx = tid + 256 * it;            // 0..1023 x 16B (B: 32x128)
            int row = idx >> 5, c4 = idx & 31;
            async_ld16(lds_off(&bf32[b][row][c4 * 4]),
                       B + (size_t)(kt + row) * N + gn0 + c4 * 4);
        }
    };

    issue(0, 0);

    const int KI = K / 32;
    for (int ki = 0; ki < KI; ++ki) {
        if (ki + 1 < KI) { issue((ki + 1) * 32, (ki + 1) & 1); WAIT_ASYNC(6); }
        else             { WAIT_ASYNC(0); }
        __syncthreads();

        // convert fp32 tiles -> bf16 fragment-friendly layouts
        const int b = ki & 1;
#pragma unroll
        for (int it = 0; it < 2; ++it) {
            int idx = tid + 256 * it;
            int row = idx >> 3, c4 = idx & 7;
            float4 av = *(const float4*)&af32[b][row][c4 * 4];
            lds_a[row][c4 * 2 + 0] = pack2(av.x, av.y);
            lds_a[row][c4 * 2 + 1] = pack2(av.z, av.w);
        }
#pragma unroll
        for (int it = 0; it < 4; ++it) {
            int idx = tid + 256 * it;
            int row = idx >> 5, c4 = idx & 31;
            float4 bv = *(const float4*)&bf32[b][row][c4 * 4];
            int n = c4 * 4;
            lds_bt[n + 0][row] = f2bf(bv.x);
            lds_bt[n + 1][row] = f2bf(bv.y);
            lds_bt[n + 2][row] = f2bf(bv.z);
            lds_bt[n + 3][row] = f2bf(bv.w);
        }
        __syncthreads();

        // A fragment: lane M = m0+(lane&15); K halves per ISA layout
        BF16x16 af;
        const uint4* arow = (const uint4*)&lds_a[m0 + (lane & 15)][0];
        af.q[0] = arow[lo ? 0 : 1];
        af.q[1] = arow[lo ? 2 : 3];

#pragma unroll
        for (int i = 0; i < 4; ++i) {
            BF16x16 bf;
            const uint4* brow = (const uint4*)&lds_bt[n0 + i * 16 + (lane & 15)][0];
            bf.q[0] = brow[lo ? 0 : 1];
            bf.q[1] = brow[lo ? 2 : 3];
            acc[i].v = __builtin_amdgcn_wmma_f32_16x16x32_bf16(
                false, af.v, false, bf.v, (short)0, acc[i].v, false, false);
        }
    }

#pragma unroll
    for (int i = 0; i < 4; ++i)
#pragma unroll
        for (int r = 0; r < 8; ++r) {
            int row = gm0 + m0 + r + ((lane >> 4) << 3);
            int col = gn0 + n0 + i * 16 + (lane & 15);
            C[(size_t)row * N + col] = acc[i].f[r];
        }
}

// ---------------------------------------------------------------------------
// K2: RoPE on q (f32 [s][4096]) -> bf16 [head][s][128]
// ---------------------------------------------------------------------------
__global__ __launch_bounds__(256)
void psdpa_rope_pack(const float* __restrict__ q, unsigned short* __restrict__ qb) {
    int idx = blockIdx.x * 256 + threadIdx.x;     // 512*32*64 pairs
    if (idx >= S_TOT * NHEADS * 64) return;
    int d = idx & 63;
    int head = (idx >> 6) & 31;
    int s = idx >> 11;
    float inv = __powf(10000.0f, -(float)d * (1.0f / 64.0f));
    float ang = (float)s * inv;
    float c = __cosf(ang), sn = __sinf(ang);
    const float* qr = q + (size_t)s * HID + head * HD;
    float x1 = qr[d], x2 = qr[d + 64];
    unsigned short* o = qb + ((size_t)head * S_TOT + s) * HD;
    o[d]      = f2bf(x1 * c - x2 * sn);
    o[d + 64] = f2bf(x2 * c + x1 * sn);
}

// ---------------------------------------------------------------------------
// K3: flash attention over paged KV, async double-buffered K/V chunks.
// grid = (S/16, NKVH); 128 thr = 4 waves = 4 q-heads of this kv-head.
// ---------------------------------------------------------------------------
__global__ __launch_bounds__(128)
void psdpa_flash_attn(const unsigned short* __restrict__ qb,
                      const float* __restrict__ kv,
                      const int* __restrict__ btab,
                      float* __restrict__ ctx) {
    __shared__ __align__(16) float          kf32[2][32][128];  // raw K chunks
    __shared__ __align__(16) float          vf32[2][32][128];  // raw V chunks
    __shared__ __align__(16) unsigned       lds_k[32][64];     // K bf16 row-major
    __shared__ __align__(16) unsigned short lds_vt[128][32];   // V bf16 transposed
    __shared__ __align__(16) unsigned short lds_p[4][16][32];  // per-wave P tile

    const int tid  = threadIdx.x;
    const int lane = tid & 31;
    const int wave = tid >> 5;
    const bool lo  = lane < 16;
    const int kvh  = blockIdx.y;
    const int head = kvh * 4 + wave;
    const int s0   = blockIdx.x * 16;

    // preload Q fragments: A layout, 4 frags over d
    BF16x16 qf[4];
    {
        const uint4* qrow =
            (const uint4*)(qb + ((size_t)head * S_TOT + s0 + (lane & 15)) * HD);
#pragma unroll
        for (int f = 0; f < 4; ++f) {
            qf[f].q[0] = qrow[f * 4 + (lo ? 0 : 1)];
            qf[f].q[1] = qrow[f * 4 + (lo ? 2 : 3)];
        }
    }

    F32x8 O[8];
#pragma unroll
    for (int f = 0; f < 8; ++f)
#pragma unroll
        for (int r = 0; r < 8; ++r) O[f].f[r] = 0.0f;
    float mrow[8], lrow[8];
#pragma unroll
    for (int r = 0; r < 8; ++r) { mrow[r] = -INFINITY; lrow[r] = 0.0f; }

    const float scale = 0.08838834764831843f;  // 1/sqrt(128)

    // issue K/V async loads for chunk cc into buffer b (16 async / wave)
    auto issue = [&](int cc, int b) {
#pragma unroll
        for (int it = 0; it < 8; ++it) {
            int idx = tid + 128 * it;            // 0..1023 x 16B
            int r = idx >> 5, c4 = idx & 31;
            int blk = btab[cc * 2 + (r >> 4)];
            const float* kp = kv + (size_t)blk * 32768 + (size_t)kvh * 2048
                                 + (r & 15) * 128 + c4 * 4;
            async_ld16(lds_off(&kf32[b][r][c4 * 4]), kp);
            async_ld16(lds_off(&vf32[b][r][c4 * 4]), kp + 16384);
        }
    };

    issue(0, 0);

    for (int c = 0; c < NCHUNK; ++c) {
        if (c + 1 < NCHUNK) { issue(c + 1, (c + 1) & 1); WAIT_ASYNC(16); }
        else                { WAIT_ASYNC(0); }
        __syncthreads();

        // convert fp32 chunk -> bf16 (K row-major, V transposed)
        const int b = c & 1;
#pragma unroll
        for (int it = 0; it < 8; ++it) {
            int idx = tid + 128 * it;
            int r = idx >> 5, c4 = idx & 31;
            float4 kval = *(const float4*)&kf32[b][r][c4 * 4];
            lds_k[r][c4 * 2 + 0] = pack2(kval.x, kval.y);
            lds_k[r][c4 * 2 + 1] = pack2(kval.z, kval.w);
            float4 vval = *(const float4*)&vf32[b][r][c4 * 4];
            int n = c4 * 4;
            lds_vt[n + 0][r] = f2bf(vval.x);
            lds_vt[n + 1][r] = f2bf(vval.y);
            lds_vt[n + 2][r] = f2bf(vval.z);
            lds_vt[n + 3][r] = f2bf(vval.w);
        }
        __syncthreads();

        // scores: S(16x32) = Q(16x128) . K^T, 2 N-frags x 4 K-steps
        F32x8 sc[2];
#pragma unroll
        for (int tf = 0; tf < 2; ++tf) {
#pragma unroll
            for (int r = 0; r < 8; ++r) sc[tf].f[r] = 0.0f;
            const uint4* krow = (const uint4*)&lds_k[tf * 16 + (lane & 15)][0];
#pragma unroll
            for (int f = 0; f < 4; ++f) {
                BF16x16 bf;
                bf.q[0] = krow[f * 4 + (lo ? 0 : 1)];
                bf.q[1] = krow[f * 4 + (lo ? 2 : 3)];
                sc[tf].v = __builtin_amdgcn_wmma_f32_16x16x32_bf16(
                    false, qf[f].v, false, bf.v, (short)0, sc[tf].v, false, false);
            }
        }

        // online softmax; row of element r = r + 8*(lane>=16)
#pragma unroll
        for (int r = 0; r < 8; ++r) {
            float x0 = sc[0].f[r] * scale;
            float x1 = sc[1].f[r] * scale;
            float mx = fmaxf(x0, x1);
#pragma unroll
            for (int msk = 1; msk <= 8; msk <<= 1)
                mx = fmaxf(mx, __shfl_xor(mx, msk, 32));
            float mnew = fmaxf(mrow[r], mx);
            float alpha = __expf(mrow[r] - mnew);
            float p0 = __expf(x0 - mnew);
            float p1 = __expf(x1 - mnew);
            float rs = p0 + p1;
#pragma unroll
            for (int msk = 1; msk <= 8; msk <<= 1)
                rs += __shfl_xor(rs, msk, 32);
            lrow[r] = lrow[r] * alpha + rs;
            mrow[r] = mnew;
#pragma unroll
            for (int f = 0; f < 8; ++f) O[f].f[r] *= alpha;
            int prow = r + ((lane >> 4) << 3);
            int pcol = lane & 15;
            lds_p[wave][prow][pcol]      = f2bf(p0);
            lds_p[wave][prow][pcol + 16] = f2bf(p1);
        }

        // P C-layout -> A-layout via LDS round-trip (wave-private region)
        BF16x16 pf;
        const uint4* prow = (const uint4*)&lds_p[wave][lane & 15][0];
        pf.q[0] = prow[lo ? 0 : 1];
        pf.q[1] = prow[lo ? 2 : 3];

        // ctx += P(16x32) @ V(32x128)
#pragma unroll
        for (int df = 0; df < 8; ++df) {
            BF16x16 vf;
            const uint4* vrow = (const uint4*)&lds_vt[df * 16 + (lane & 15)][0];
            vf.q[0] = vrow[lo ? 0 : 1];
            vf.q[1] = vrow[lo ? 2 : 3];
            O[df].v = __builtin_amdgcn_wmma_f32_16x16x32_bf16(
                false, pf.v, false, vf.v, (short)0, O[df].v, false, false);
        }
    }

    // normalize + store ctx as f32 [s][4096]
#pragma unroll
    for (int df = 0; df < 8; ++df)
#pragma unroll
        for (int r = 0; r < 8; ++r) {
            int s = s0 + r + ((lane >> 4) << 3);
            int col = head * HD + df * 16 + (lane & 15);
            ctx[(size_t)s * HID + col] = O[df].f[r] / lrow[r];
        }
}

// ---------------------------------------------------------------------------
extern "C" void kernel_launch(void* const* d_in, const int* in_sizes, int n_in,
                              void* d_out, int out_size, void* d_ws, size_t ws_size,
                              hipStream_t stream) {
    const float* hidden = (const float*)d_in[0];
    const float* Wq     = (const float*)d_in[1];
    const float* Wo     = (const float*)d_in[2];
    const float* kv     = (const float*)d_in[3];
    const int*   btab   = (const int*)d_in[4];

    char* ws = (char*)d_ws;
    float*          q_ws   = (float*)ws;                               // 8 MB
    unsigned short* qb_ws  = (unsigned short*)(ws + 8u * 1024 * 1024); // 4 MB
    float*          ctx_ws = (float*)(ws + 12u * 1024 * 1024);         // 8 MB

    // 1) q = hidden @ Wq
    psdpa_gemm_bf16<<<dim3(HID / 128, S_TOT / 64), 256, 0, stream>>>(
        hidden, Wq, q_ws, S_TOT, HID, HID);
    // 2) RoPE + pack to bf16 [head][s][128]
    psdpa_rope_pack<<<(S_TOT * NHEADS * 64 + 255) / 256, 256, 0, stream>>>(q_ws, qb_ws);
    // 3) paged flash attention
    psdpa_flash_attn<<<dim3(S_TOT / 16, NKVH), 128, 0, stream>>>(qb_ws, kv, btab, ctx_ws);
    // 4) out = ctx @ Wo
    psdpa_gemm_bf16<<<dim3(HID / 128, S_TOT / 64), 256, 0, stream>>>(
        ctx_ws, Wo, (float*)d_out, S_TOT, HID, HID);
}
